// SALayer_33354716020952
// MI455X (gfx1250) — compile-verified, matrix operand order checked by reading, not run on hardware
//
#include <hip/hip_runtime.h>
#include <cstdint>

typedef __attribute__((ext_vector_type(16))) _Float16 v16h;
typedef __attribute__((ext_vector_type(8)))  _Float16 v8h;
typedef __attribute__((ext_vector_type(8)))  float    v8f;

#define NB   16
#define NP   4096
#define NC   64
#define NS   1024
#define NK   32
#define R2   0.04f
#define M_TOT (NB * NS * NK)          // 524288
#define M_TILES (M_TOT / 16)          // 32768

// ---------------------------------------------------------------------------
// Weight prep: f32 [rows, cols] -> f16 [rows, kpad] (zero padded)
// ---------------------------------------------------------------------------
__global__ void prep_w_kernel(const float* __restrict__ W, _Float16* __restrict__ Wp,
                              int rows, int cols, int kpad) {
    int i = blockIdx.x * blockDim.x + threadIdx.x;
    if (i < rows * kpad) {
        int r = i / kpad, c = i % kpad;
        Wp[i] = (_Float16)(c < cols ? W[r * cols + c] : 0.0f);
    }
}

__global__ void zero_kernel(float* __restrict__ p, int n) {
    int i = blockIdx.x * blockDim.x + threadIdx.x;
    if (i < n) p[i] = 0.0f;
}

// ---------------------------------------------------------------------------
// Farthest point sampling: one block per batch, 16 points/thread in registers.
// wave32 shuffle argmax, then tiny 8-way LDS combine (3 barriers / round).
// ---------------------------------------------------------------------------
#define FPS_T 256
#define PPT   (NP / FPS_T)   // 16

__global__ void fps_kernel(const float* __restrict__ xyz,
                           int* __restrict__ sidx,
                           float* __restrict__ new_xyz) {
    int b = blockIdx.x;
    int t = threadIdx.x;
    int lane = t & 31, wave = t >> 5;
    const float* base = xyz + (size_t)b * NP * 3;

    float px[PPT], py[PPT], pz[PPT], dst[PPT];
#pragma unroll
    for (int i = 0; i < PPT; i++) {
        int p = t * PPT + i;
        px[i] = base[p * 3 + 0];
        py[i] = base[p * 3 + 1];
        pz[i] = base[p * 3 + 2];
        dst[i] = 1e10f;
    }

    __shared__ float wrv[FPS_T / 32];
    __shared__ int   wri[FPS_T / 32];
    __shared__ float cent[3];
    __shared__ int   bestIdx;

    int farthest = 0;
    for (int it = 0; it < NS; it++) {
        if (farthest >= t * PPT && farthest < (t + 1) * PPT) {
            int l = farthest - t * PPT;
            cent[0] = px[l]; cent[1] = py[l]; cent[2] = pz[l];
        }
        __syncthreads();
        if (t == 0) {
            sidx[b * NS + it] = farthest;
            new_xyz[((size_t)b * NS + it) * 3 + 0] = cent[0];
            new_xyz[((size_t)b * NS + it) * 3 + 1] = cent[1];
            new_xyz[((size_t)b * NS + it) * 3 + 2] = cent[2];
        }
        float cx = cent[0], cy = cent[1], cz = cent[2];
        float bm = -1.0f; int bi = 0;
#pragma unroll
        for (int i = 0; i < PPT; i++) {
            float dx = px[i] - cx, dy = py[i] - cy, dz = pz[i] - cz;
            float d = dx * dx + dy * dy + dz * dz;
            dst[i] = fminf(dst[i], d);
            if (dst[i] > bm) { bm = dst[i]; bi = t * PPT + i; }
        }
        // wave32 shuffle argmax
#pragma unroll
        for (int off = 16; off > 0; off >>= 1) {
            float ov = __shfl_xor(bm, off, 32);
            int   oi = __shfl_xor(bi, off, 32);
            if (ov > bm) { bm = ov; bi = oi; }
        }
        if (lane == 0) { wrv[wave] = bm; wri[wave] = bi; }
        __syncthreads();
        if (t == 0) {
            float gb = wrv[0]; int gi = wri[0];
#pragma unroll
            for (int w = 1; w < FPS_T / 32; w++)
                if (wrv[w] > gb) { gb = wrv[w]; gi = wri[w]; }
            bestIdx = gi;
        }
        __syncthreads();
        farthest = bestIdx;
    }
}

// ---------------------------------------------------------------------------
// Ball query (32-NN, radius-masked to nearest) + gather + concat -> f16
// grouped[M, 96]: [0:3) = xyz - center, [3:67) = feat, [67:96) = 0 pad.
// ---------------------------------------------------------------------------
#define GQ_T 128

__global__ void group_kernel(const float* __restrict__ xyz,
                             const float* __restrict__ feat,
                             const float* __restrict__ new_xyz,
                             _Float16* __restrict__ grouped) {
    int bs = blockIdx.x;          // 0 .. NB*NS-1
    int b  = bs >> 10;
    int t  = threadIdx.x;
    int lane = t & 31, wave = t >> 5;

    __shared__ float d2[NP];
    __shared__ float wrv[GQ_T / 32];
    __shared__ int   wri[GQ_T / 32];
    __shared__ int   selIdx[NK];
    __shared__ float selD[NK];

    const float* base = xyz + (size_t)b * NP * 3;
    float cx = new_xyz[bs * 3 + 0];
    float cy = new_xyz[bs * 3 + 1];
    float cz = new_xyz[bs * 3 + 2];

    for (int p = t; p < NP; p += GQ_T) {
        float dx = base[p * 3 + 0] - cx;
        float dy = base[p * 3 + 1] - cy;
        float dz = base[p * 3 + 2] - cz;
        d2[p] = dx * dx + dy * dy + dz * dz;
    }
    __syncthreads();

    for (int k = 0; k < NK; k++) {
        float bm = 1e30f; int bi = 0;
        for (int p = t; p < NP; p += GQ_T) {
            float d = d2[p];
            if (d < bm) { bm = d; bi = p; }
        }
#pragma unroll
        for (int off = 16; off > 0; off >>= 1) {
            float ov = __shfl_xor(bm, off, 32);
            int   oi = __shfl_xor(bi, off, 32);
            if (ov < bm) { bm = ov; bi = oi; }
        }
        if (lane == 0) { wrv[wave] = bm; wri[wave] = bi; }
        __syncthreads();
        if (t == 0) {
            float gb = wrv[0]; int gi = wri[0];
#pragma unroll
            for (int w = 1; w < GQ_T / 32; w++)
                if (wrv[w] < gb) { gb = wrv[w]; gi = wri[w]; }
            selIdx[k] = gi;
            selD[k]   = gb;
            d2[gi]    = 1e30f;
        }
        __syncthreads();
    }
    if (t > 0 && t < NK && selD[t] > R2) selIdx[t] = selIdx[0];
    __syncthreads();

    const float* fbase = feat + (size_t)b * NP * NC;
    _Float16* gbase = grouped + (size_t)bs * NK * 96;
    for (int e = t; e < NK * 96; e += GQ_T) {
        int k = e / 96, c = e % 96;
        int p = selIdx[k];
        float v;
        if (c < 3)       v = base[p * 3 + c] - new_xyz[bs * 3 + c];
        else if (c < 67) v = fbase[p * NC + (c - 3)];
        else             v = 0.0f;
        gbase[e] = (_Float16)v;
    }
}

// ---------------------------------------------------------------------------
// WMMA GEMM: Y[M, NT*16] = act(A)[M,KPAD] x Wp[NT*16,KPAD]^T + bias.
// One wave owns one 16-row M tile and ALL NT output column tiles, so A is
// read exactly once from HBM. Optional fused input BN+ReLU (scale/shift of
// the PREVIOUS layer applied while building A fragments). f32 accumulate,
// f16 store, per-channel sum/sum^2 for this layer's batch-norm stats.
// ---------------------------------------------------------------------------
template <int KPAD, int NT, bool FUSE>
__global__ void gemm_bn_kernel(const _Float16* __restrict__ A,
                               const _Float16* __restrict__ Wp,
                               const float* __restrict__ bias,
                               const float* __restrict__ scshA,  // [KPAD]=scale,[KPAD..2K)=shift
                               _Float16* __restrict__ Y,
                               float* __restrict__ sums) {       // [0,Nch)=sum,[Nch,2Nch)=sumsq
    constexpr int Nch = NT * 16;
    int lane  = threadIdx.x & 31;
    int wave  = threadIdx.x >> 5;
    int mtile = blockIdx.x * 8 + wave;
    int hs    = lane >> 4;        // 0 -> K 0..7/16..23 ; 1 -> 8..15/24..31
    int lm    = lane & 15;
    int base0 = hs * 8;

    __shared__ float s_sc[KPAD], s_sh[KPAD];
    __shared__ float lsum[Nch], lsq[Nch];
    if (FUSE) {
        for (int i = threadIdx.x; i < KPAD; i += blockDim.x) {
            s_sc[i] = scshA[i];
            s_sh[i] = scshA[KPAD + i];
        }
    }
    if (threadIdx.x < Nch) { lsum[threadIdx.x] = 0.0f; lsq[threadIdx.x] = 0.0f; }
    __syncthreads();

    const _Float16* arow = A + (size_t)(mtile * 16 + lm) * KPAD;

    v8f acc[NT];
#pragma unroll
    for (int nt = 0; nt < NT; nt++) acc[nt] = v8f{};

#pragma unroll
    for (int kt = 0; kt < KPAD; kt += 32) {
        v8h alo = *(const v8h*)(arow + kt + base0);
        v8h ahi = *(const v8h*)(arow + kt + base0 + 16);
        v16h a;
        if (FUSE) {
#pragma unroll
            for (int i = 0; i < 8; i++) {
                int klo = kt + base0 + i;
                int khi = kt + base0 + 16 + i;
                float vlo = fmaxf((float)alo[i] * s_sc[klo] + s_sh[klo], 0.0f);
                float vhi = fmaxf((float)ahi[i] * s_sc[khi] + s_sh[khi], 0.0f);
                a[i]     = (_Float16)vlo;
                a[i + 8] = (_Float16)vhi;
            }
        } else {
#pragma unroll
            for (int i = 0; i < 8; i++) { a[i] = alo[i]; a[i + 8] = ahi[i]; }
        }
#pragma unroll
        for (int nt = 0; nt < NT; nt++) {
            const _Float16* brow = Wp + (size_t)(nt * 16 + lm) * KPAD;
            v8h blo = *(const v8h*)(brow + kt + base0);
            v8h bhi = *(const v8h*)(brow + kt + base0 + 16);
            v16h bm;
#pragma unroll
            for (int i = 0; i < 8; i++) { bm[i] = blo[i]; bm[i + 8] = bhi[i]; }
            acc[nt] = __builtin_amdgcn_wmma_f32_16x16x32_f16(
                false, a, false, bm, (short)0, acc[nt], false, false);
        }
    }

#pragma unroll
    for (int nt = 0; nt < NT; nt++) {
        int n = nt * 16 + lm;
        float bn = bias[n];
        float s = 0.0f, q = 0.0f;
#pragma unroll
        for (int r = 0; r < 8; r++) {
            int m = mtile * 16 + hs * 8 + r;   // D layout: VGPR r -> M = r + 8*hs
            float v = acc[nt][r] + bn;
            Y[(size_t)m * Nch + n] = (_Float16)v;
            s += v; q += v * v;
        }
        atomicAdd(&lsum[n], s);
        atomicAdd(&lsq[n], q);
    }
    __syncthreads();
    if (threadIdx.x < Nch) {
        atomicAdd(&sums[threadIdx.x],       lsum[threadIdx.x]);
        atomicAdd(&sums[Nch + threadIdx.x], lsq[threadIdx.x]);
    }
}

// ---------------------------------------------------------------------------
// BN: stats -> per-channel scale/shift
// ---------------------------------------------------------------------------
__global__ void bn_scale_kernel(const float* __restrict__ sums,
                                const float* __restrict__ g,
                                const float* __restrict__ be,
                                float* __restrict__ scsh, int Nch, float invM) {
    int c = threadIdx.x;
    if (c < Nch) {
        float mean = sums[c] * invM;
        float var  = sums[Nch + c] * invM - mean * mean;
        float sc   = g[c] * rsqrtf(var + 1e-5f);
        scsh[c]        = sc;
        scsh[Nch + c]  = be[c] - mean * sc;
    }
}

// ---------------------------------------------------------------------------
// Final: BN+ReLU of layer2 fused with max over K -> f32 output [B,S,128]
// ---------------------------------------------------------------------------
__global__ void max_kernel(const _Float16* __restrict__ Y2,
                           const float* __restrict__ scsh,
                           float* __restrict__ out) {
    int bs = blockIdx.x;
    int c  = threadIdx.x;   // 128
    const _Float16* p = Y2 + (size_t)bs * NK * 128 + c;
    float sc = scsh[c], sh = scsh[128 + c];
    float m = 0.0f;   // relu output >= 0
#pragma unroll 4
    for (int k = 0; k < NK; k++) {
        float v = fmaxf((float)p[(size_t)k * 128] * sc + sh, 0.0f);
        m = fmaxf(m, v);
    }
    out[(size_t)bs * 128 + c] = m;
}

// ---------------------------------------------------------------------------
extern "C" void kernel_launch(void* const* d_in, const int* in_sizes, int n_in,
                              void* d_out, int out_size, void* d_ws, size_t ws_size,
                              hipStream_t stream) {
    const float* xyz  = (const float*)d_in[0];
    const float* feat = (const float*)d_in[1];
    const float* W0 = (const float*)d_in[2];
    const float* b0 = (const float*)d_in[3];
    const float* g0 = (const float*)d_in[4];
    const float* be0 = (const float*)d_in[5];
    const float* W1 = (const float*)d_in[6];
    const float* b1 = (const float*)d_in[7];
    const float* g1 = (const float*)d_in[8];
    const float* be1 = (const float*)d_in[9];
    const float* W2 = (const float*)d_in[10];
    const float* b2 = (const float*)d_in[11];
    const float* g2 = (const float*)d_in[12];
    const float* be2 = (const float*)d_in[13];

    float* out = (float*)d_out;
    float* out_newxyz = out;                       // [B,S,3]
    float* out_feat   = out + (size_t)NB * NS * 3; // [B,S,128]

    // --- workspace layout (256B aligned) ---
    char* ws = (char*)d_ws;
    size_t off = 0;
    auto take = [&](size_t bytes) {
        off = (off + 255) & ~(size_t)255;
        char* p = ws + off;
        off += bytes;
        return p;
    };
    int*      sidx    = (int*)     take((size_t)NB * NS * sizeof(int));
    _Float16* grouped = (_Float16*)take((size_t)M_TOT * 96  * sizeof(_Float16));
    _Float16* y0      = (_Float16*)take((size_t)M_TOT * 64  * sizeof(_Float16));
    _Float16* y2      = (_Float16*)take((size_t)M_TOT * 128 * sizeof(_Float16));
    _Float16* Wp0     = (_Float16*)take(64  * 96 * sizeof(_Float16));
    _Float16* Wp1     = (_Float16*)take(64  * 64 * sizeof(_Float16));
    _Float16* Wp2     = (_Float16*)take(128 * 64 * sizeof(_Float16));
    float*    sums    = (float*)   take(512 * sizeof(float));  // 128 + 128 + 256
    float*    scsh    = (float*)   take(512 * sizeof(float));
    _Float16* y1 = grouped;   // reuse: grouped dead after gemm0

    float* sums0 = sums;        float* scsh0 = scsh;
    float* sums1 = sums + 128;  float* scsh1 = scsh + 128;
    float* sums2 = sums + 256;  float* scsh2 = scsh + 256;

    const float invM = 1.0f / (float)M_TOT;

    // weights -> f16 padded [N, Kpad]; zero stat accumulators
    prep_w_kernel<<<(64 * 96 + 255) / 256, 256, 0, stream>>>(W0, Wp0, 64, 67, 96);
    prep_w_kernel<<<(64 * 64 + 255) / 256, 256, 0, stream>>>(W1, Wp1, 64, 64, 64);
    prep_w_kernel<<<(128 * 64 + 255) / 256, 256, 0, stream>>>(W2, Wp2, 128, 64, 64);
    zero_kernel<<<2, 256, 0, stream>>>(sums, 512);

    // sampling + grouping
    fps_kernel<<<NB, FPS_T, 0, stream>>>(xyz, sidx, out_newxyz);
    group_kernel<<<NB * NS, GQ_T, 0, stream>>>(xyz, feat, out_newxyz, grouped);

    // layer 0: [M,96] x [96,64]        (raw grouped input)
    gemm_bn_kernel<96, 4, false><<<M_TILES / 8, 256, 0, stream>>>(
        grouped, Wp0, b0, nullptr, y0, sums0);
    bn_scale_kernel<<<1, 64, 0, stream>>>(sums0, g0, be0, scsh0, 64, invM);

    // layer 1: [M,64] x [64,64]        (BN0+ReLU fused into A read)
    gemm_bn_kernel<64, 4, true><<<M_TILES / 8, 256, 0, stream>>>(
        y0, Wp1, b1, scsh0, y1, sums1);
    bn_scale_kernel<<<1, 64, 0, stream>>>(sums1, g1, be1, scsh1, 64, invM);

    // layer 2: [M,64] x [64,128]       (BN1+ReLU fused into A read)
    gemm_bn_kernel<64, 8, true><<<M_TILES / 8, 256, 0, stream>>>(
        y1, Wp2, b2, scsh1, y2, sums2);
    bn_scale_kernel<<<1, 128, 0, stream>>>(sums2, g2, be2, scsh2, 128, invM);

    // fused BN2+ReLU+max over nsample -> f32 out
    max_kernel<<<NB * NS, 128, 0, stream>>>(y2, scsh2, out_feat);

    (void)in_sizes; (void)n_in; (void)out_size; (void)ws_size;
}
